// RateRNN_5909874999707
// MI455X (gfx1250) — compile-verified, matrix operand order checked by reading, not run on hardware
//
#include <hip/hip_runtime.h>

// ---------------------------------------------------------------------------
// RateRNN on MI455X (gfx1250, wave32, WMMA)
//
// u_{t+1} = clip(u + DT*tau_inv*(f + b + W_rec@relu(u) + eta - u))
// eta_{t+1} = eta*decay + nscale*z
//
// Per-batch workgroup (8 WGs x 1024 threads = 32 waves). Wave w owns M-tile w
// (rows w*16..w*16+15) and both N-tiles (trials 0..15, 16..31). u/eta stay in
// registers for all 400 steps; relu(u) is re-published to LDS (split bf16
// hi/lo, transposed [n][k], pitch 520 for bank-conflict-free b128 access)
// every step. GEMM = 3-term split-bf16 WMMA (Whi*rhi + Whi*rlo + Wlo*rhi),
// f32 accumulate => ~fp32 accuracy at bf16 matrix throughput.
// ---------------------------------------------------------------------------

typedef __attribute__((ext_vector_type(16))) __bf16 v16bf;
typedef __attribute__((ext_vector_type(8)))  __bf16 v8bf;
typedef __attribute__((ext_vector_type(8)))  float  v8f;

#define N_NEURONS  512
#define N_EXC      410
#define N_INPUT    128
#define NUM_TRIALS 32
#define BATCH      8
#define NUM_STEPS  400
#define LDS_PITCH  520   // 512 + 8 pad -> 1040B rows (4-bank skew, 16B aligned)

static __device__ __forceinline__ v16bf cat8(v8bf a, v8bf b) {
  return __builtin_shufflevector(a, b, 0,1,2,3,4,5,6,7,8,9,10,11,12,13,14,15);
}

// ---------------------------------------------------------------------------
// Pre-swizzle W_rec (fp32 512x512 row-major) into the WMMA 16-bit A-operand
// lane layout, split into bf16 hi + bf16 lo (lo = fp32 residual).
// Tile (mt,kt) occupies 512 bf16 elements: chunk0[lane]*8 .. chunk1[lane]*8,
// where lane = (m%16) + 16*half, chunk0 holds K = kt*32 + half*8 + 0..7 and
// chunk1 holds K = kt*32 + 16 + half*8 + 0..7 (per CDNA5 ISA §7.12.2).
// ---------------------------------------------------------------------------
__global__ __launch_bounds__(256) void pack_wrec(const float* __restrict__ W,
                                                 __bf16* __restrict__ Ahi,
                                                 __bf16* __restrict__ Alo) {
  const int idx = blockIdx.x * blockDim.x + threadIdx.x;  // 0..32767
  const int m  = idx >> 6;          // row 0..511
  const int kc = (idx & 63) << 3;   // 8-wide k chunk start
  const float* src = W + m * N_NEURONS + kc;
  v8bf hi, lo;
#pragma unroll
  for (int i = 0; i < 8; ++i) {
    float wv = src[i];
    __bf16 h = (__bf16)wv;
    hi[i] = h;
    lo[i] = (__bf16)(wv - (float)h);
  }
  const int mt = m >> 4, ml = m & 15;
  const int kt = kc >> 5, rem = kc & 31;
  const int half = (rem & 15) >> 3;   // 0: K+0..7 band, 1: K+8..15 band
  const int ch   = rem >> 4;          // 0: chunk0 (K+0..7/8..15), 1: chunk1 (K+16..)
  const int lane = ml + (half << 4);
  const int dst  = (mt * 16 + kt) * 512 + ch * 256 + lane * 8;
  *(v8bf*)(Ahi + dst) = hi;
  *(v8bf*)(Alo + dst) = lo;
}

// fb[b][n] = relu(W_in[n,:] . h[b,:]) + bias[n]
__global__ __launch_bounds__(512) void input_proj(const float* __restrict__ Win,
                                                  const float* __restrict__ h,
                                                  const float* __restrict__ bias,
                                                  float* __restrict__ fb) {
  __shared__ float sh[N_INPUT];
  const int b = blockIdx.x;
  const int t = threadIdx.x;
  if (t < N_INPUT) sh[t] = h[b * N_INPUT + t];
  __syncthreads();
  const float* wr = Win + t * N_INPUT;
  float acc = 0.0f;
#pragma unroll 4
  for (int i = 0; i < N_INPUT; i += 4)
    acc += wr[i] * sh[i] + wr[i + 1] * sh[i + 1] +
           wr[i + 2] * sh[i + 2] + wr[i + 3] * sh[i + 3];
  acc = acc > 0.0f ? acc : 0.0f;
  fb[b * N_NEURONS + t] = acc + bias[t];
}

// ---------------------------------------------------------------------------
// Main time-stepping kernel: one block per batch element.
// ---------------------------------------------------------------------------
__global__ __launch_bounds__(1024) void rnn_kernel(
    const float* __restrict__ u0, const float* __restrict__ eta0,
    const float* __restrict__ noise, const __bf16* __restrict__ Ahi,
    const __bf16* __restrict__ Alo, const float* __restrict__ fb,
    float* __restrict__ hist, float* __restrict__ etaout) {
  // r transposed [n][k], split hi/lo. 2 * 32 * 520 * 2B = 66,560 B of LDS.
  __shared__ __bf16 s_hi[NUM_TRIALS][LDS_PITCH];
  __shared__ __bf16 s_lo[NUM_TRIALS][LDS_PITCH];

  const int b     = blockIdx.x;
  const int tid   = threadIdx.x;
  const int lane  = tid & 31;
  const int w     = tid >> 5;          // wave id == M-tile (0..31)
  const int half  = lane >> 4;         // lane half selects row group / K band
  const int nl    = lane & 15;         // column within a 16-wide N tile
  const int mbase = w * 16 + half * 8; // 8 consecutive rows owned by this lane

  constexpr float DTE   = 0.0005f / 0.02f;             // DT/TAU_E
  constexpr float DTI   = 0.0005f / 0.01f;             // DT/TAU_I
  constexpr float DECAY = 1.0f - 0.0005f / 0.02f;      // 1 - DT/TAU_ETA
  constexpr float NSCALE = 0.1f * 0.223606797749979f;  // SIGMA*sqrt(2*DT/TAU)

  float u[2][8], eta[2][8], fbv[8], dtt[8];
#pragma unroll
  for (int i = 0; i < 8; ++i) {
    const int m = mbase + i;
    fbv[i] = fb[b * N_NEURONS + m];
    dtt[i] = (m < N_EXC) ? DTE : DTI;
  }
#pragma unroll
  for (int nt = 0; nt < 2; ++nt) {
    const int n = nt * 16 + nl;
#pragma unroll
    for (int i = 0; i < 8; ++i) {
      const int idx = (b * N_NEURONS + mbase + i) * NUM_TRIALS + n;
      u[nt][i]   = u0[idx];
      eta[nt][i] = eta0[idx];
    }
  }

  // Publish relu(u) split-bf16, transposed, in the WMMA B-operand layout.
  auto publish = [&]() {
#pragma unroll
    for (int nt = 0; nt < 2; ++nt) {
      const int n = nt * 16 + nl;
      v8bf hi8, lo8;
#pragma unroll
      for (int i = 0; i < 8; ++i) {
        float r = u[nt][i] > 0.0f ? u[nt][i] : 0.0f;
        __bf16 rh = (__bf16)r;
        hi8[i] = rh;
        lo8[i] = (__bf16)(r - (float)rh);
      }
      *(v8bf*)(&s_hi[n][mbase]) = hi8;  // one ds_store_b128 each
      *(v8bf*)(&s_lo[n][mbase]) = lo8;
    }
  };

  publish();
  __syncthreads();

  for (int step = 0; step < NUM_STEPS; ++step) {
    // Prefetch this step's noise (streaming -> non-temporal).
    const int sb = step * BATCH + b;
    const float* np = noise + (size_t)sb * N_NEURONS * NUM_TRIALS;
    float z[2][8];
#pragma unroll
    for (int nt = 0; nt < 2; ++nt) {
      const int n = nt * 16 + nl;
#pragma unroll
      for (int i = 0; i < 8; ++i)
        z[nt][i] = __builtin_nontemporal_load(np + (mbase + i) * NUM_TRIALS + n);
    }

    v8f acc0 = {};  // (mt=w, nt=0)
    v8f acc1 = {};  // (mt=w, nt=1)

#pragma unroll 4
    for (int kt = 0; kt < 16; ++kt) {
      const int tbase = (w * 16 + kt) * 512;  // pre-swizzled A tile
      v16bf Ah = cat8(*(const v8bf*)(Ahi + tbase + lane * 8),
                      *(const v8bf*)(Ahi + tbase + 256 + lane * 8));
      v16bf Al = cat8(*(const v8bf*)(Alo + tbase + lane * 8),
                      *(const v8bf*)(Alo + tbase + 256 + lane * 8));

      const int k0 = kt * 32 + half * 8;
      v16bf Bh0 = cat8(*(const v8bf*)&s_hi[nl][k0],      *(const v8bf*)&s_hi[nl][k0 + 16]);
      v16bf Bl0 = cat8(*(const v8bf*)&s_lo[nl][k0],      *(const v8bf*)&s_lo[nl][k0 + 16]);
      v16bf Bh1 = cat8(*(const v8bf*)&s_hi[16 + nl][k0], *(const v8bf*)&s_hi[16 + nl][k0 + 16]);
      v16bf Bl1 = cat8(*(const v8bf*)&s_lo[16 + nl][k0], *(const v8bf*)&s_lo[16 + nl][k0 + 16]);

      // Split GEMM: Whi*rhi + Whi*rlo + Wlo*rhi (f32 accumulate)
      acc0 = __builtin_amdgcn_wmma_f32_16x16x32_bf16(false, Ah, false, Bh0, (short)0, acc0, false, false);
      acc0 = __builtin_amdgcn_wmma_f32_16x16x32_bf16(false, Ah, false, Bl0, (short)0, acc0, false, false);
      acc0 = __builtin_amdgcn_wmma_f32_16x16x32_bf16(false, Al, false, Bh0, (short)0, acc0, false, false);
      acc1 = __builtin_amdgcn_wmma_f32_16x16x32_bf16(false, Ah, false, Bh1, (short)0, acc1, false, false);
      acc1 = __builtin_amdgcn_wmma_f32_16x16x32_bf16(false, Ah, false, Bl1, (short)0, acc1, false, false);
      acc1 = __builtin_amdgcn_wmma_f32_16x16x32_bf16(false, Al, false, Bh1, (short)0, acc1, false, false);
    }

    // Elementwise update + history store (non-temporal: pure streaming).
    float* op = hist + (size_t)sb * N_NEURONS * NUM_TRIALS;
#pragma unroll
    for (int nt = 0; nt < 2; ++nt) {
      const int n = nt * 16 + nl;
#pragma unroll
      for (int i = 0; i < 8; ++i) {
        const float rec = nt ? acc1[i] : acc0[i];
        const float uu  = u[nt][i];
        float un = uu + dtt[i] * (fbv[i] + rec + eta[nt][i] - uu);
        un = fminf(20.0f, fmaxf(-5.0f, un));
        eta[nt][i] = eta[nt][i] * DECAY + NSCALE * z[nt][i];
        u[nt][i] = un;
        __builtin_nontemporal_store(un, op + (mbase + i) * NUM_TRIALS + n);
      }
    }

    __syncthreads();  // all waves done reading r before it is overwritten
    publish();
    __syncthreads();  // new r visible to all waves
  }

#pragma unroll
  for (int nt = 0; nt < 2; ++nt) {
    const int n = nt * 16 + nl;
#pragma unroll
    for (int i = 0; i < 8; ++i)
      etaout[(b * N_NEURONS + mbase + i) * NUM_TRIALS + n] = eta[nt][i];
  }
}

extern "C" void kernel_launch(void* const* d_in, const int* in_sizes, int n_in,
                              void* d_out, int out_size, void* d_ws, size_t ws_size,
                              hipStream_t stream) {
  const float* u0    = (const float*)d_in[0];
  const float* h     = (const float*)d_in[1];
  const float* eta0  = (const float*)d_in[2];
  const float* noise = (const float*)d_in[3];
  const float* W_rec = (const float*)d_in[4];
  const float* W_in  = (const float*)d_in[5];
  const float* bias  = (const float*)d_in[6];

  // Workspace: A_hi (512KB) | A_lo (512KB) | fb (16KB)
  __bf16* Ahi = (__bf16*)d_ws;
  __bf16* Alo = Ahi + N_NEURONS * N_NEURONS;
  float*  fbw = (float*)(Alo + N_NEURONS * N_NEURONS);

  float* hist   = (float*)d_out;
  float* etaout = hist + (size_t)NUM_STEPS * BATCH * N_NEURONS * NUM_TRIALS;

  pack_wrec <<<dim3(128),   dim3(256), 0, stream>>>(W_rec, Ahi, Alo);
  input_proj<<<dim3(BATCH), dim3(512), 0, stream>>>(W_in, h, bias, fbw);
  rnn_kernel<<<dim3(BATCH), dim3(1024), 0, stream>>>(u0, eta0, noise, Ahi, Alo,
                                                     fbw, hist, etaout);
}
